// ScoreNetwork_46600395162217
// MI455X (gfx1250) — matrix operand order repaired and checked by reading, not compiled, not run on hardware
//
#include <hip/hip_runtime.h>
#include <hip/hip_bf16.h>

// ---------------------------------------------------------------------------
// Types for CDNA5 WMMA (wave32, 16x16x32 f16 -> f32)
// ---------------------------------------------------------------------------
typedef __attribute__((ext_vector_type(16))) _Float16 v16h;
typedef __attribute__((ext_vector_type(8)))  _Float16 v8h;
typedef __attribute__((ext_vector_type(8)))  float    v8f;
typedef __attribute__((ext_vector_type(4)))  int      v4i;
typedef __attribute__((ext_vector_type(8)))  int      v8i;

// A-fragment (16xK tile, row-major in LDS, stride in halfs).
// lane<16 -> M=lane, K=0..15 contiguous; lane>=16 -> K=16..31 (matches frag_b)
__device__ __forceinline__ v16h frag_a(const _Float16* base, int rowBase, int stride, int lane) {
    int r   = rowBase + (lane & 15);
    int klo = (lane >> 4) << 4;
    const _Float16* p = base + r * stride + klo;
    union { v16h v; v8h h[2]; } u;
    u.h[0] = *(const v8h*)(p);
    u.h[1] = *(const v8h*)(p + 8);
    return u.v;
}

// B-fragment from B^T stored row-major [n][k] in LDS.
// k order used here (matches frag_a): lane<16 -> N=lane, K=0..15 contiguous; lane>=16 -> K=16..31
__device__ __forceinline__ v16h frag_b(const _Float16* base, int rowBase, int stride, int lane) {
    int n  = rowBase + (lane & 15);
    int kb = (lane >> 4) << 4;
    const _Float16* p = base + n * stride + kb;
    union { v16h v; v8h h[2]; } u;
    u.h[0] = *(const v8h*)(p);
    u.h[1] = *(const v8h*)(p + 8);
    return u.v;
}

__device__ __forceinline__ v8f wmma16(v16h a, v16h b, v8f c) {
    return __builtin_amdgcn_wmma_f32_16x16x32_f16(false, a, false, b, (short)0, c, false, false);
}

// ---------------------------------------------------------------------------
// Tensor Data Mover: 2-D f16 tile (rows x cols) global -> LDS (compacted).
// D# per CDNA5 ISA 8.3/8.4. Issued by one wave; completion via TENSORcnt.
// ---------------------------------------------------------------------------
__device__ __forceinline__ void tdm_load_2d_f16(const _Float16* gptr, void* lds,
                                                unsigned strideElems,
                                                unsigned rows, unsigned cols,
                                                unsigned dim0, unsigned dim1)
{
    unsigned long long ga = (unsigned long long)(uintptr_t)gptr;
    unsigned ldsOff = (unsigned)(uintptr_t)lds;   // low 32 bits of generic = LDS offset
    v4i g0;
    g0[0] = 1;                                    // count=1 (valid user descriptor)
    g0[1] = (int)ldsOff;                          // lds_addr
    g0[2] = (int)(unsigned)(ga & 0xffffffffu);    // global_addr[31:0]
    g0[3] = (int)(((unsigned)(ga >> 32) & 0x01ffffffu) | (2u << 30)); // addr[56:32] | type=2
    v8i g1;
    g1[0] = (1 << 16);                            // data_size=1 (2 bytes); no multicast
    g1[1] = (int)((dim0 & 0xffffu) << 16);        // tensor_dim0[15:0]
    g1[2] = (int)(((dim0 >> 16) & 0xffffu) | ((dim1 & 0xffffu) << 16));
    g1[3] = (int)(((dim1 >> 16) & 0xffffu) | ((cols & 0xffffu) << 16)); // tile_dim0
    g1[4] = (int)(rows & 0xffffu);                // tile_dim1 (tile_dim2 = 0 -> 2D)
    g1[5] = (int)strideElems;                     // tensor_dim0_stride[31:0]
    g1[6] = 0;
    g1[7] = 0;
    v4i gz = {0, 0, 0, 0};                        // groups 2/3: zero (dims unused)
    asm volatile("tensor_load_to_lds %0, %1, %2, %3"
                 :: "s"(g0), "s"(g1), "s"(gz), "s"(gz)
                 : "memory");
}

// ---------------------------------------------------------------------------
// WMMA GEMM: C[M,N] = act(A16[M,K] @ WT[N,K]^T + bias + rowAdd)
// A16: f16 row-major [M][K]; WT: f16 pre-transposed weights [N][K].
// WG tile 64(M) x 128(N), 8 waves of 32x32, K-step 64 -> 8 WMMAs per barrier.
// Tiles moved global->LDS by the Tensor Data Mover. Optional f16 output copy.
// ---------------------------------------------------------------------------
__global__ __launch_bounds__(256) void gemm_wmma_kernel(
    const _Float16* __restrict__ A16, const _Float16* __restrict__ WT,
    const float* __restrict__ bias, const float* __restrict__ rowAdd,
    float* __restrict__ C, _Float16* __restrict__ C16,
    int M, int K, int N, int act, int rowsPerBatch)
{
    __shared__ __attribute__((aligned(32))) _Float16 Ah[64 * 64];    // 8 KB
    __shared__ __attribute__((aligned(32))) _Float16 Bt[128 * 64];   // 16 KB

    const int tid  = threadIdx.x;
    const int lane = tid & 31;
    const int wave = tid >> 5;
    const int m0 = blockIdx.x * 64;
    const int n0 = blockIdx.y * 128;
    const int waveM = (wave & 1) * 32;
    const int waveN = (wave >> 1) * 32;

    v8f c00 = {}, c01 = {}, c10 = {}, c11 = {};

    for (int k0 = 0; k0 < K; k0 += 64) {
        __syncthreads();                     // previous tiles fully consumed
        if (tid < 32) {                      // wave 0 drives the TDM (EXEC all-ones)
            tdm_load_2d_f16(A16 + (size_t)m0 * K + k0, Ah, K, 64, 64,
                            (unsigned)(K - k0), 64u);
            tdm_load_2d_f16(WT + (size_t)n0 * K + k0, Bt, K, 128, 64,
                            (unsigned)(K - k0), 128u);
            __builtin_amdgcn_s_wait_tensorcnt(0);
        }
        __syncthreads();                     // tiles visible to all waves

#pragma unroll
        for (int c = 0; c < 2; ++c) {        // two 32-K chunks
            const _Float16* Ac = Ah + c * 32;
            const _Float16* Bc = Bt + c * 32;
            v16h a0 = frag_a(Ac, waveM,      64, lane);
            v16h a1 = frag_a(Ac, waveM + 16, 64, lane);
            v16h b0 = frag_b(Bc, waveN,      64, lane);
            v16h b1 = frag_b(Bc, waveN + 16, 64, lane);
            c00 = wmma16(a0, b0, c00);
            c01 = wmma16(a0, b1, c01);
            c10 = wmma16(a1, b0, c10);
            c11 = wmma16(a1, b1, c11);
        }
    }

    // Store per C layout: VGPR v -> M = v + 8*(lane>=16), N = lane&15
    const int rb   = (lane >> 4) << 3;
    const int col0 = n0 + waveN + (lane & 15);
    const int col1 = col0 + 16;
    float bias0 = bias ? bias[col0] : 0.f;
    float bias1 = bias ? bias[col1] : 0.f;
#pragma unroll
    for (int v = 0; v < 8; ++v) {
        int rows[2] = { m0 + waveM + rb + v, m0 + waveM + rb + v + 16 };
        float vals[2][2] = { { c00[v], c01[v] }, { c10[v], c11[v] } };
#pragma unroll
        for (int rr = 0; rr < 2; ++rr) {
            int row = rows[rr];
            float v0 = vals[rr][0] + bias0;
            float v1 = vals[rr][1] + bias1;
            if (rowAdd) {
                int bidx = row / rowsPerBatch;
                v0 += rowAdd[(size_t)bidx * N + col0];
                v1 += rowAdd[(size_t)bidx * N + col1];
            }
            if (act) { v0 = fmaxf(v0, 0.f); v1 = fmaxf(v1, 0.f); }
            C[(size_t)row * N + col0] = v0;
            C[(size_t)row * N + col1] = v1;
            if (C16) {
                C16[(size_t)row * N + col0] = (_Float16)v0;
                C16[(size_t)row * N + col1] = (_Float16)v1;
            }
        }
    }
}

// ---------------------------------------------------------------------------
// Weight prep: W[K][N] f32 -> WT[N][K] f16 (transposed).  idx sweeps W linearly.
// ---------------------------------------------------------------------------
__global__ __launch_bounds__(256) void transpose_cvt_kernel(
    const float* __restrict__ W, _Float16* __restrict__ WT, int K, int N)
{
    int idx = blockIdx.x * 256 + threadIdx.x;
    if (idx < K * N) {
        int k = idx / N, n = idx - k * N;
        WT[(size_t)n * K + k] = (_Float16)W[idx];
    }
}

__global__ __launch_bounds__(256) void cvt_f16_kernel(
    const float* __restrict__ src, _Float16* __restrict__ dst, int n)
{
    int idx = blockIdx.x * 256 + threadIdx.x;
    if (idx < n) dst[idx] = (_Float16)src[idx];
}

// ---------------------------------------------------------------------------
// Fused attention, one block per (batch, head). N=200 (pad 208/224), DH=32.
// Scores in registers (13 x v8f), softmax via half-group shuffles, P->LDS f16,
// PV WMMAs. Writes f32 + f16 copies of O.
// ---------------------------------------------------------------------------
__global__ __launch_bounds__(128) void attn_kernel(
    const float* __restrict__ qkv, float* __restrict__ obuf,
    _Float16* __restrict__ obuf16)
{
    __shared__ __attribute__((aligned(32))) _Float16 Kh[208 * 32];
    __shared__ __attribute__((aligned(32))) _Float16 Vh[32 * 224];
    __shared__ __attribute__((aligned(32))) _Float16 Ph[4][16 * 224];
    __shared__ __attribute__((aligned(32))) _Float16 Qh[4][16 * 32];

    const int bh = blockIdx.x, b = bh >> 3, hh = bh & 7;
    const int tid = threadIdx.x, lane = tid & 31, wave = tid >> 5;
    const size_t rowBase = (size_t)b * 200;

    for (int idx = tid; idx < 208 * 32; idx += 128) {
        int n = idx >> 5, d = idx & 31;
        Kh[idx] = (n < 200) ? (_Float16)qkv[(rowBase + n) * 768 + 256 + hh * 32 + d]
                            : (_Float16)0.0f;
    }
    for (int idx = tid; idx < 32 * 224; idx += 128) {
        int d = idx / 224, n = idx - d * 224;
        Vh[idx] = (n < 200) ? (_Float16)qkv[(rowBase + n) * 768 + 512 + hh * 32 + d]
                            : (_Float16)0.0f;
    }
    __syncthreads();

    const float scale = 0.17677669529663687f;  // 1/sqrt(32)

    for (int qt = wave; qt < 13; qt += 4) {
        const int m0 = qt * 16;
        for (int j = 0; j < 16; ++j) {
            int idx = j * 32 + lane;
            int r = idx >> 5, d = idx & 31;
            int n = m0 + r;
            Qh[wave][idx] = (n < 200) ? (_Float16)qkv[(rowBase + n) * 768 + hh * 32 + d]
                                      : (_Float16)0.0f;
        }
        v16h aq = frag_a(Qh[wave], 0, 32, lane);

        v8f s[13];
#pragma unroll
        for (int nt = 0; nt < 13; ++nt) {
            v16h bk = frag_b(Kh, nt * 16, 32, lane);
            v8f z = {};
            s[nt] = wmma16(aq, bk, z);
        }

        const int colb = lane & 15;
        float mx[8], sm[8];
#pragma unroll
        for (int v = 0; v < 8; ++v) mx[v] = -1e30f;
#pragma unroll
        for (int nt = 0; nt < 13; ++nt) {
            bool ok = (nt * 16 + colb) < 200;
#pragma unroll
            for (int v = 0; v < 8; ++v) {
                float val = ok ? s[nt][v] * scale : -1e30f;
                s[nt][v] = val;
                mx[v] = fmaxf(mx[v], val);
            }
        }
#pragma unroll
        for (int v = 0; v < 8; ++v) {
            for (int off = 1; off < 16; off <<= 1)
                mx[v] = fmaxf(mx[v], __shfl_xor(mx[v], off, 32));
            sm[v] = 0.f;
        }
#pragma unroll
        for (int nt = 0; nt < 13; ++nt)
#pragma unroll
            for (int v = 0; v < 8; ++v) {
                float e = __expf(s[nt][v] - mx[v]);
                s[nt][v] = e;
                sm[v] += e;
            }
#pragma unroll
        for (int v = 0; v < 8; ++v) {
            for (int off = 1; off < 16; off <<= 1)
                sm[v] += __shfl_xor(sm[v], off, 32);
            sm[v] = 1.f / sm[v];
        }

        const int rb = (lane >> 4) << 3;
#pragma unroll
        for (int nt = 0; nt < 13; ++nt)
#pragma unroll
            for (int v = 0; v < 8; ++v)
                Ph[wave][(rb + v) * 224 + nt * 16 + colb] = (_Float16)(s[nt][v] * sm[v]);
        for (int idx = lane; idx < 256; idx += 32) {
            int r = idx >> 4, c = idx & 15;
            Ph[wave][r * 224 + 208 + c] = (_Float16)0.0f;
        }

        v8f o0 = {}, o1 = {};
#pragma unroll
        for (int kc = 0; kc < 7; ++kc) {
            v16h ap  = frag_a(Ph[wave] + kc * 32, 0,  224, lane);
            v16h bv0 = frag_b(Vh + kc * 32,       0,  224, lane);
            v16h bv1 = frag_b(Vh + kc * 32,       16, 224, lane);
            o0 = wmma16(ap, bv0, o0);
            o1 = wmma16(ap, bv1, o1);
        }
        const int d0 = lane & 15;
#pragma unroll
        for (int v = 0; v < 8; ++v) {
            int n = m0 + rb + v;
            if (n < 200) {
                size_t p0 = (rowBase + n) * 256 + hh * 32 + d0;
                obuf[p0]        = o0[v];
                obuf[p0 + 16]   = o1[v];
                obuf16[p0]      = (_Float16)o0[v];
                obuf16[p0 + 16] = (_Float16)o1[v];
            }
        }
    }
}

// ---------------------------------------------------------------------------
// x = LayerNorm(x + y) * g + b (in place), plus f16 copy.
// ---------------------------------------------------------------------------
__global__ __launch_bounds__(256) void add_ln_kernel(
    float* __restrict__ x, const float* __restrict__ y,
    const float* __restrict__ g, const float* __restrict__ bb,
    _Float16* __restrict__ x16)
{
    const int row = blockIdx.x, tid = threadIdx.x;
    __shared__ float red[256];
    float v = x[(size_t)row * 256 + tid] + y[(size_t)row * 256 + tid];
    red[tid] = v; __syncthreads();
    for (int s = 128; s > 0; s >>= 1) { if (tid < s) red[tid] += red[tid + s]; __syncthreads(); }
    float mean = red[0] * (1.f / 256.f);
    __syncthreads();
    float d = v - mean;
    red[tid] = d * d; __syncthreads();
    for (int s = 128; s > 0; s >>= 1) { if (tid < s) red[tid] += red[tid + s]; __syncthreads(); }
    float var = red[0] * (1.f / 256.f);
    float o = d * rsqrtf(var + 1e-5f) * g[tid] + bb[tid];
    x[(size_t)row * 256 + tid] = o;
    x16[(size_t)row * 256 + tid] = (_Float16)o;
}

// ---------------------------------------------------------------------------
// t_emb[b] = silu(t[b]*tw1 + tb1) @ tw2 + tb2.
// ---------------------------------------------------------------------------
__global__ __launch_bounds__(256) void time_embed_kernel(
    const float* __restrict__ t, const float* __restrict__ tw1,
    const float* __restrict__ tb1, const float* __restrict__ tw2,
    const float* __restrict__ tb2, float* __restrict__ temb)
{
    const int b = blockIdx.x, j = threadIdx.x;
    __shared__ float s[256];
    float u = t[b] * tw1[j] + tb1[j];
    s[j] = u / (1.f + __expf(-u));
    __syncthreads();
    float acc = tb2[j];
    for (int i = 0; i < 256; ++i) acc += s[i] * tw2[i * 256 + j];
    temb[b * 256 + j] = acc;
}

// ---------------------------------------------------------------------------
// Rank-1 collapse of edge_emb @ wc (ob1 folded into the constant term).
// ---------------------------------------------------------------------------
__global__ __launch_bounds__(256) void cvec_kernel(
    const float* __restrict__ ew, const float* __restrict__ eb,
    const float* __restrict__ wc, const float* __restrict__ ob1,
    float* __restrict__ cvec)
{
    const int j = threadIdx.x;
    float v1 = 0.f, v0 = 0.f;
    for (int i = 0; i < 256; ++i) {
        float w = wc[i * 256 + j];
        v1 += ew[i] * w;
        v0 += eb[i] * w;
    }
    cvec[j]       = v1;
    cvec[256 + j] = v0 + ob1[j];
}

// ---------------------------------------------------------------------------
// Pairwise output: out[b,i,j,:] = silu(A[b,i]+Bv[b,j]+xt*v1+v0) @ ow2 + ob2
// ---------------------------------------------------------------------------
__global__ __launch_bounds__(256) void pairwise_kernel(
    const float* __restrict__ Ap, const float* __restrict__ Bp,
    const float* __restrict__ xt, const float* __restrict__ cvec,
    const float* __restrict__ ow2, const float* __restrict__ ob2,
    float* __restrict__ out)
{
    const int blk = blockIdx.x, b = blk / 200, i = blk - b * 200;
    const int tid = threadIdx.x, lane = tid & 31, wave = tid >> 5;
    __shared__ float Ar[256], V1[256], V0[256], W0[256], W1[256];
    Ar[tid] = Ap[((size_t)b * 200 + i) * 256 + tid];
    V1[tid] = cvec[tid];
    V0[tid] = cvec[256 + tid];
    W0[tid] = ow2[tid * 2];
    W1[tid] = ow2[tid * 2 + 1];
    __syncthreads();
    const float o0b = ob2[0], o1b = ob2[1];

    for (int j = wave; j < 200; j += 8) {
        float xv = xt[((size_t)b * 200 + i) * 200 + j];
        const float* brow = Bp + ((size_t)b * 200 + j) * 256;
        if (j + 8 < 200)
            __builtin_prefetch(Bp + ((size_t)b * 200 + j + 8) * 256 + lane * 8, 0, 1);
        float a0 = 0.f, a1 = 0.f;
        for (int hh = lane; hh < 256; hh += 32) {
            float pre = Ar[hh] + brow[hh] + xv * V1[hh] + V0[hh];
            float sv  = pre / (1.f + __expf(-pre));
            a0 += sv * W0[hh];
            a1 += sv * W1[hh];
        }
        for (int off = 16; off > 0; off >>= 1) {
            a0 += __shfl_xor(a0, off, 32);
            a1 += __shfl_xor(a1, off, 32);
        }
        if (lane == 0) {
            size_t o = (((size_t)b * 200 + i) * 200 + j) * 2;
            out[o]     = a0 + o0b;
            out[o + 1] = a1 + o1b;
        }
    }
}

// ---------------------------------------------------------------------------
// Launch
// ---------------------------------------------------------------------------
extern "C" void kernel_launch(void* const* d_in, const int* in_sizes, int n_in,
                              void* d_out, int out_size, void* d_ws, size_t ws_size,
                              hipStream_t stream) {
    const float* h     = (const float*)d_in[0];
    const float* xt    = (const float*)d_in[1];
    const float* t     = (const float*)d_in[2];
    const float* tw1   = (const float*)d_in[3];
    const float* tb1   = (const float*)d_in[4];
    const float* tw2   = (const float*)d_in[5];
    const float* tb2   = (const float*)d_in[6];
    const float* ew    = (const float*)d_in[7];
    const float* eb    = (const float*)d_in[8];
    const float* pw    = (const float*)d_in[9];
    const float* pb    = (const float*)d_in[10];
    const float* qkv_w = (const float*)d_in[11];
    const float* qkv_b = (const float*)d_in[12];
    const float* ao_w  = (const float*)d_in[13];
    const float* ao_b  = (const float*)d_in[14];
    const float* ln1_g = (const float*)d_in[15];
    const float* ln1_b = (const float*)d_in[16];
    const float* f1_w  = (const float*)d_in[17];
    const float* f1_b  = (const float*)d_in[18];
    const float* f2_w  = (const float*)d_in[19];
    const float* f2_b  = (const float*)d_in[20];
    const float* ln2_g = (const float*)d_in[21];
    const float* ln2_b = (const float*)d_in[22];
    const float* ow1   = (const float*)d_in[23];
    const float* ob1   = (const float*)d_in[24];
    const float* ow2   = (const float*)d_in[25];
    const float* ob2   = (const float*)d_in[26];
    float* out = (float*)d_out;

    // ---- f32 workspace ----
    float* ws   = (float*)d_ws;
    float* X    = ws;                 // 1600*256
    float* TMP  = ws + 409600;        // 1600*256
    float* QKV  = ws + 819200;        // 1600*768
    float* OBUF = ws + 2048000;       // 1600*256
    float* FF   = ws + 2457600;       // 1600*1024
    float* TEMB = ws + 4096000;       // 8*256
    float* APJ  = ws + 4098048;       // 1600*256
    float* BPJ  = ws + 4507648;       // 1600*256
    float* CV   = ws + 4917248;       // 2*256
    // ---- f16 workspace (after f32 region, 16B aligned) ----
    _Float16* hws  = (_Float16*)(ws + 4917760);
    _Float16* H16  = hws;             // 1600*128
    _Float16* X16  = hws + 204800;    // 1600*256
    _Float16* O16  = hws + 614400;    // 1600*256
    _Float16* FF16 = hws + 1024000;   // 1600*1024
    _Float16* WPW  = hws + 2662400;   // 256*128
    _Float16* WQKV = hws + 2695168;   // 3 * 768*256
    _Float16* WAO  = hws + 3284992;   // 3 * 256*256
    _Float16* WF1  = hws + 3481600;   // 3 * 1024*256
    _Float16* WF2  = hws + 4268032;   // 3 * 256*1024
    _Float16* WWA  = hws + 5054464;   // 256*256
    _Float16* WWB  = hws + 5120000;   // 256*256

    // ---- prep: f16 activations input + transposed f16 weights ----
    cvt_f16_kernel<<<800, 256, 0, stream>>>(h, H16, 204800);
    transpose_cvt_kernel<<<128, 256, 0, stream>>>(pw, WPW, 128, 256);
    for (int l = 0; l < 3; ++l) {
        transpose_cvt_kernel<<<768, 256, 0, stream>>>(qkv_w + (size_t)l * 196608,
                                                      WQKV + (size_t)l * 196608, 256, 768);
        transpose_cvt_kernel<<<256, 256, 0, stream>>>(ao_w + (size_t)l * 65536,
                                                      WAO + (size_t)l * 65536, 256, 256);
        transpose_cvt_kernel<<<1024, 256, 0, stream>>>(f1_w + (size_t)l * 262144,
                                                       WF1 + (size_t)l * 262144, 256, 1024);
        transpose_cvt_kernel<<<1024, 256, 0, stream>>>(f2_w + (size_t)l * 262144,
                                                       WF2 + (size_t)l * 262144, 1024, 256);
    }
    transpose_cvt_kernel<<<256, 256, 0, stream>>>(ow1,         WWA, 256, 256);
    transpose_cvt_kernel<<<256, 256, 0, stream>>>(ow1 + 65536, WWB, 256, 256);

    // ---- forward ----
    time_embed_kernel<<<8, 256, 0, stream>>>(t, tw1, tb1, tw2, tb2, TEMB);
    gemm_wmma_kernel<<<dim3(25, 2), 256, 0, stream>>>(H16, WPW, pb, TEMB, X, X16,
                                                      1600, 128, 256, 0, 200);
    for (int l = 0; l < 3; ++l) {
        gemm_wmma_kernel<<<dim3(25, 6), 256, 0, stream>>>(
            X16, WQKV + (size_t)l * 196608, qkv_b + l * 768, nullptr,
            QKV, nullptr, 1600, 256, 768, 0, 1);
        attn_kernel<<<64, 128, 0, stream>>>(QKV, OBUF, O16);
        gemm_wmma_kernel<<<dim3(25, 2), 256, 0, stream>>>(
            O16, WAO + (size_t)l * 65536, ao_b + l * 256, nullptr,
            TMP, nullptr, 1600, 256, 256, 0, 1);
        add_ln_kernel<<<1600, 256, 0, stream>>>(X, TMP, ln1_g + l * 256, ln1_b + l * 256, X16);
        gemm_wmma_kernel<<<dim3(25, 8), 256, 0, stream>>>(
            X16, WF1 + (size_t)l * 262144, f1_b + l * 1024, nullptr,
            FF, FF16, 1600, 256, 1024, 1, 1);
        gemm_wmma_kernel<<<dim3(25, 2), 256, 0, stream>>>(
            FF16, WF2 + (size_t)l * 262144, f2_b + l * 256, nullptr,
            TMP, nullptr, 1600, 1024, 256, 0, 1);
        add_ln_kernel<<<1600, 256, 0, stream>>>(X, TMP, ln2_g + l * 256, ln2_b + l * 256, X16);
    }
    gemm_wmma_kernel<<<dim3(25, 2), 256, 0, stream>>>(X16, WWA, nullptr, nullptr,
                                                      APJ, nullptr, 1600, 256, 256, 0, 1);
    gemm_wmma_kernel<<<dim3(25, 2), 256, 0, stream>>>(X16, WWB, nullptr, nullptr,
                                                      BPJ, nullptr, 1600, 256, 256, 0, 1);
    cvec_kernel<<<1, 256, 0, stream>>>(ew, eb, ow1 + 131072, ob1, CV);
    pairwise_kernel<<<1600, 256, 0, stream>>>(APJ, BPJ, xt, CV, ow2, ob2, out);
}
